// LoRAAttention_5480378269787
// MI455X (gfx1250) — matrix operand order repaired and checked by reading, not compiled, not run on hardware
//
#include <hip/hip_runtime.h>
#include <cstdint>

typedef unsigned short u16;
typedef __attribute__((ext_vector_type(16))) __bf16 v16bf;
typedef __attribute__((ext_vector_type(8)))  __bf16 v8bf;
typedef __attribute__((ext_vector_type(8)))  float  v8f;
typedef __attribute__((ext_vector_type(4)))  unsigned int uint32x4;
typedef __attribute__((ext_vector_type(8)))  int int32x8;
typedef __attribute__((ext_vector_type(4)))  int int32x4;

// exact pointee type for the async-LDS builtins: gcc-style vector of 4 ints
typedef int v4i __attribute__((vector_size(16)));
typedef v4i __attribute__((address_space(1))) global_v4i;
typedef v4i __attribute__((address_space(3))) shared_v4i;

#define BDIM 8
#define NSEQ 1024
#define CDIM 768
#define NH   12
#define HD   64
#define RANK 64
#define M_TOTAL (BDIM * NSEQ)      /* 8192 */
#define LDS_STRIDE 40              /* 32 + 8 pad, keeps 16B alignment */

#if defined(__has_builtin)
#if __has_builtin(__builtin_amdgcn_global_load_async_to_lds_b128)
#define HAVE_ASYNC_LDS 1
#endif
#if __has_builtin(__builtin_amdgcn_tensor_load_to_lds) && __has_builtin(__builtin_amdgcn_s_wait_tensorcnt)
#define HAVE_TDM 1
#endif
#endif

__device__ __forceinline__ u16 f2bf(float f) {
    union { float f; unsigned u; } v; v.f = f;
    unsigned u = v.u;
    u += 0x7FFFu + ((u >> 16) & 1u);   // round-to-nearest-even
    return (u16)(u >> 16);
}

__device__ __forceinline__ v8f wmma_bf16(v16bf a, v16bf b, v8f c) {
    return __builtin_amdgcn_wmma_f32_16x16x32_bf16(
        false, a, false, b, (short)0, c, false, false);
}

union bf16x16 { v16bf v; v8bf h[2]; };

// A fragment (16x32, M x K): lane l -> row m=l&15; e<8 -> K=half*8+e, e>=8 -> K=16+half*8+(e-8)
__device__ __forceinline__ v16bf frag_A(const u16* lds, int stride, int row0, int k0) {
    const int l = threadIdx.x & 31;
    const int m = l & 15, half = l >> 4;
    const u16* p = lds + (row0 + m) * stride + k0 + half * 8;
    bf16x16 u;
    u.h[0] = *(const v8bf*)(const void*)(p);
    u.h[1] = *(const v8bf*)(const void*)(p + 16);
    return u.v;
}

// B fragment (32x16, K x N) sourced from row-major [n][k]: lane l -> n=l&15; e<8 -> K=half*8+e, e>=8 -> K=16+half*8+(e-8)
__device__ __forceinline__ v16bf frag_B(const u16* lds, int stride, int row0, int k0) {
    const int l = threadIdx.x & 31;
    const int n = l & 15, half = l >> 4;
    const u16* p = lds + (row0 + n) * stride + k0 + half * 8;
    bf16x16 u;
    u.h[0] = *(const v8bf*)(const void*)(p);
    u.h[1] = *(const v8bf*)(const void*)(p + 16);
    return u.v;
}

// 16-byte global -> LDS copy: async on CDNA5 (ASYNCcnt), VGPR round-trip otherwise
__device__ __forceinline__ void cp16_g2l(u16* ldst, const u16* gsrc) {
#if defined(HAVE_ASYNC_LDS)
    __builtin_amdgcn_global_load_async_to_lds_b128(
        (global_v4i*)(uintptr_t)gsrc,
        (shared_v4i*)(uintptr_t)(unsigned)(uintptr_t)ldst,  // low 32 bits = LDS offset
        0, 0);
#else
    *(uint4*)ldst = *(const uint4*)(const void*)gsrc;
#endif
}

__device__ __forceinline__ void async_wait() {
#if defined(HAVE_ASYNC_LDS)
#if __has_builtin(__builtin_amdgcn_s_wait_asynccnt)
    __builtin_amdgcn_s_wait_asynccnt(0);
#else
    asm volatile("s_wait_asynccnt 0x0" ::: "memory");
#endif
#endif
}

// cooperative 128x32 bf16 tile issue, 256 threads, 8-elem chunks
__device__ __forceinline__ void g2l_tile(u16* dst, const u16* src, int src_stride,
                                         long row0, int k0, int tid) {
#pragma unroll
    for (int it = 0; it < 2; ++it) {
        int c   = tid + it * 256;
        int r   = c >> 2;
        int col = (c & 3) << 3;
        const u16* sp = src + (long)(row0 + r) * src_stride + k0 + col;
        __builtin_prefetch(sp + 64, 0, 3);   // near-cache prefetch, 2 K-tiles ahead
        cp16_g2l(dst + r * LDS_STRIDE + col, sp);
    }
}

// ---------------------------------------------------------------- f32 -> bf16
__global__ __launch_bounds__(256) void k_cvt(const float* __restrict__ src,
                                             u16* __restrict__ dst, int n) {
    for (int i = blockIdx.x * blockDim.x + threadIdx.x; i < n; i += gridDim.x * blockDim.x)
        dst[i] = f2bf(src[i]);
}

// ------------------------------------------------- t = x @ [A_k;A_v]^T * (alpha/rank)
__device__ __forceinline__ void lorat_loadB(u16* dst, const u16* Ak, const u16* Av,
                                            int k0, int tid) {
#pragma unroll
    for (int it = 0; it < 2; ++it) {
        int c   = tid + it * 256;
        int r   = c >> 2;
        int col = (c & 3) << 3;
        const u16* src = (r < 64) ? (Ak + (long)r * CDIM) : (Av + (long)(r - 64) * CDIM);
        cp16_g2l(dst + r * LDS_STRIDE + col, src + k0 + col);
    }
}

__global__ __launch_bounds__(256) void k_lorat(const u16* __restrict__ xb,
                                               const u16* __restrict__ Ak,
                                               const u16* __restrict__ Av,
                                               u16* __restrict__ tb) {
    const int tid = threadIdx.x;
    const int lanei = tid & 31;
    const int wave = tid >> 5;
    const int wm = wave & 3, wn = wave >> 2;
    const long tileM = (long)blockIdx.x * 128;

    __shared__ u16 sA[2][128 * LDS_STRIDE];
    __shared__ u16 sB[2][128 * LDS_STRIDE];

    v8f z = {};
    v8f acc[2][4];
#pragma unroll
    for (int i = 0; i < 2; ++i)
#pragma unroll
        for (int j = 0; j < 4; ++j) acc[i][j] = z;

    g2l_tile(sA[0], xb, CDIM, tileM, 0, tid);
    lorat_loadB(sB[0], Ak, Av, 0, tid);
    async_wait();
    __syncthreads();

    int cur = 0;
    for (int k0 = 0; k0 < CDIM; k0 += 32) {
        if (k0 + 32 < CDIM) {                       // issue next tile while computing
            g2l_tile(sA[cur ^ 1], xb, CDIM, tileM, k0 + 32, tid);
            lorat_loadB(sB[cur ^ 1], Ak, Av, k0 + 32, tid);
        }
        v16bf a0 = frag_A(sA[cur], LDS_STRIDE, wm * 32, 0);
        v16bf a1 = frag_A(sA[cur], LDS_STRIDE, wm * 32 + 16, 0);
#pragma unroll
        for (int ni = 0; ni < 4; ++ni) {
            v16bf bfr = frag_B(sB[cur], LDS_STRIDE, wn * 64 + ni * 16, 0);
            acc[0][ni] = wmma_bf16(a0, bfr, acc[0][ni]);
            acc[1][ni] = wmma_bf16(a1, bfr, acc[1][ni]);
        }
        async_wait();
        __syncthreads();
        cur ^= 1;
    }

    const int half = lanei >> 4, ln = lanei & 15;
#pragma unroll
    for (int mi = 0; mi < 2; ++mi)
#pragma unroll
        for (int ni = 0; ni < 4; ++ni)
#pragma unroll
            for (int r = 0; r < 8; ++r) {
                long row = tileM + wm * 32 + mi * 16 + half * 8 + r;
                int  col = wn * 64 + ni * 16 + ln;
                tb[row * 128 + col] = f2bf(acc[mi][ni][r] * (1.0f / 64.0f));
            }
}

// ------------------------- qkv = x @ qkv_w^T + b  (+ fused LoRA adds on K/V sections)
__global__ __launch_bounds__(256) void k_qkv(const u16* __restrict__ xb,
                                             const u16* __restrict__ wb,
                                             const float* __restrict__ bias,
                                             const u16* __restrict__ tb,
                                             const u16* __restrict__ Bk,
                                             const u16* __restrict__ Bv,
                                             u16* __restrict__ qbuf,
                                             u16* __restrict__ kbuf,
                                             u16* __restrict__ vbuf) {
    const int tid = threadIdx.x;
    const int lanei = tid & 31;
    const int wave = tid >> 5;
    const int wm = wave & 3, wn = wave >> 2;
    const int tileN = blockIdx.x * 128;            // 0..2303
    const long tileM = (long)blockIdx.y * 128;
    const int sec = tileN / CDIM;                  // 0=Q 1=K 2=V (uniform per block)
    const int nloc = tileN - sec * CDIM;

    __shared__ u16 sA[2][128 * LDS_STRIDE];
    __shared__ u16 sB[2][128 * LDS_STRIDE];

    v8f z = {};
    v8f acc[2][4];
#pragma unroll
    for (int i = 0; i < 2; ++i)
#pragma unroll
        for (int j = 0; j < 4; ++j) acc[i][j] = z;

    g2l_tile(sA[0], xb, CDIM, tileM, 0, tid);
    g2l_tile(sB[0], wb, CDIM, tileN, 0, tid);
    async_wait();
    __syncthreads();

    int cur = 0;
    for (int k0 = 0; k0 < CDIM; k0 += 32) {
        if (k0 + 32 < CDIM) {
            g2l_tile(sA[cur ^ 1], xb, CDIM, tileM, k0 + 32, tid);
            g2l_tile(sB[cur ^ 1], wb, CDIM, tileN, k0 + 32, tid);
        }
        v16bf a0 = frag_A(sA[cur], LDS_STRIDE, wm * 32, 0);
        v16bf a1 = frag_A(sA[cur], LDS_STRIDE, wm * 32 + 16, 0);
#pragma unroll
        for (int ni = 0; ni < 4; ++ni) {
            v16bf bfr = frag_B(sB[cur], LDS_STRIDE, wn * 64 + ni * 16, 0);
            acc[0][ni] = wmma_bf16(a0, bfr, acc[0][ni]);
            acc[1][ni] = wmma_bf16(a1, bfr, acc[1][ni]);
        }
        async_wait();
        __syncthreads();
        cur ^= 1;
    }

    if (sec > 0) {                                  // fused LoRA: += t_{k,v} @ B^T
        const u16* Bl = (sec == 1) ? Bk : Bv;
        const int coff = (sec - 1) * 64;            // t columns: K-lora 0..63, V-lora 64..127
        for (int kk = 0; kk < RANK; kk += 32) {
            g2l_tile(sA[0], tb, 128, tileM, coff + kk, tid);
            g2l_tile(sB[0], Bl, RANK, nloc, kk, tid);
            async_wait();
            __syncthreads();
            v16bf a0 = frag_A(sA[0], LDS_STRIDE, wm * 32, 0);
            v16bf a1 = frag_A(sA[0], LDS_STRIDE, wm * 32 + 16, 0);
#pragma unroll
            for (int ni = 0; ni < 4; ++ni) {
                v16bf bfr = frag_B(sB[0], LDS_STRIDE, wn * 64 + ni * 16, 0);
                acc[0][ni] = wmma_bf16(a0, bfr, acc[0][ni]);
                acc[1][ni] = wmma_bf16(a1, bfr, acc[1][ni]);
            }
            __syncthreads();
        }
    }

    // epilogue: bias, scatter into [B,H,N,HD]; fold 1/sqrt(HD) into q
    const int half = lanei >> 4, ln = lanei & 15;
#pragma unroll
    for (int mi = 0; mi < 2; ++mi)
#pragma unroll
        for (int ni = 0; ni < 4; ++ni)
#pragma unroll
            for (int r = 0; r < 8; ++r) {
                long row = tileM + wm * 32 + mi * 16 + half * 8 + r;
                int  col = tileN + wn * 64 + ni * 16 + ln;      // 0..2303
                float v = acc[mi][ni][r] + bias[col];
                int c  = col - sec * CDIM;
                int hh = c >> 6, hd = c & 63;
                long bb = row >> 10, nn = row & 1023;
                long idx = (((bb * NH + hh) * NSEQ + nn) << 6) + hd;
                if (sec == 0)      qbuf[idx] = f2bf(v * 0.125f);
                else if (sec == 1) kbuf[idx] = f2bf(v);
                else               vbuf[idx] = f2bf(v);
            }
}

// ---------------------------------------------------------------- flash attention
__global__ __launch_bounds__(256) void k_attn(const u16* __restrict__ qb,
                                              const u16* __restrict__ kb,
                                              const u16* __restrict__ vb,
                                              u16* __restrict__ aob) {
    extern __shared__ u16 smem[];
    u16* sQ  = smem;                    // 128 x 72
    u16* sK  = sQ + 128 * 72;           // 128 x 72
    u16* sVT = sK + 128 * 72;           // 64  x 136 (V transposed: [hd][kv])
    const int tid = threadIdx.x;
    const int lanei = tid & 31;
    const int wave = tid >> 5;
    u16* sP = sVT + 64 * 136 + wave * (16 * 136);  // per-wave 16 x 136 P tile

    const int bh = blockIdx.x >> 3;
    const int qblk = blockIdx.x & 7;
    const int b = bh / NH, h = bh % NH;
    const long base = (long)bh * NSEQ * HD;
    const int q0 = qblk * 128;

#pragma unroll
    for (int it = 0; it < 4; ++it) {    // q tile (scale pre-folded at store time)
        int c = tid + it * 256, r = c >> 3, col = (c & 7) << 3;
        cp16_g2l(sQ + r * 72 + col, qb + base + (long)(q0 + r) * HD + col);
    }

    float m_i[8], l_i[8];
    v8f z = {};
    v8f acc_o[4];
#pragma unroll
    for (int r = 0; r < 8; ++r) { m_i[r] = -1e30f; l_i[r] = 0.0f; }
#pragma unroll
    for (int t = 0; t < 4; ++t) acc_o[t] = z;

    const int half = lanei >> 4, ln = lanei & 15;

    for (int j = 0; j < 8; ++j) {
        const int kv0 = j * 128;
#if defined(HAVE_TDM)
        // K tile via Tensor Data Mover: 128x64 bf16 tile of the [1024,64] tensor,
        // LDS padding 4 DWORDs after every 32 DWORDs -> row stride 72 elements.
        if (tid < 32) {
            unsigned long long ga =
                (unsigned long long)(uintptr_t)(const void*)(kb + base + (long)kv0 * HD);
            unsigned lds_lo = (unsigned)(uintptr_t)(void*)sK;
            uint32x4 g0 = {0u, 0u, 0u, 0u};
            g0[0] = 1u;                                   // count=1, user descriptor
            g0[1] = lds_lo;                               // lds_addr
            g0[2] = (unsigned)ga;                         // global_addr[31:0]
            g0[3] = (unsigned)(ga >> 32) | (2u << 30);    // global_addr[56:32] | type=2
            int32x8 g1 = {0, 0, 0, 0, 0, 0, 0, 0};
            g1[0] = (1 << 16)                             // data_size = 2 bytes
                  | (1 << 20)                             // pad_enable
                  | (4 << 22)                             // pad_interval: 32 DWORDs
                  | (3 << 25);                            // pad_amount:   4 DWORDs
            g1[1] = (int)(64u << 16);                     // tensor_dim0 = 64
            g1[2] = (int)(1024u << 16);                   // tensor_dim1 = 1024
            g1[3] = (int)(64u << 16);                     // tile_dim0 = 64
            g1[4] = 128;                                  // tile_dim1 = 128
            g1[5] = 64;                                   // tensor_dim0_stride = 64
            int32x4 gz = {0, 0, 0, 0};
#if __clang_major__ >= 23
            int32x8 gz8 = {0, 0, 0, 0, 0, 0, 0, 0};
            __builtin_amdgcn_tensor_load_to_lds(g0, g1, gz, gz, gz8, 0);
#else
            __builtin_amdgcn_tensor_load_to_lds(g0, g1, gz, gz, 0);
#endif
            __builtin_amdgcn_s_wait_tensorcnt(0);
        }
#else
#pragma unroll
        for (int it = 0; it < 4; ++it) {            // K tile row-major [kv][hd]
            int c = tid + it * 256, r = c >> 3, col = (c & 7) << 3;
            cp16_g2l(sK + r * 72 + col, kb + base + (long)(kv0 + r) * HD + col);
        }
#endif
        for (int i = tid; i < 128 * HD; i += 256) { // V tile transposed -> [hd][kv]
            int kv = i >> 6, hd = i & 63;
            sVT[hd * 136 + kv] = vb[base + (long)(kv0 + kv) * HD + hd];
        }
        async_wait();
        __syncthreads();

        // S = q @ k^T (already scaled), 16 rows x 128 cols per wave
        v16bf aq0 = frag_A(sQ, 72, wave * 16, 0);
        v16bf aq1 = frag_A(sQ, 72, wave * 16, 32);
        v8f s[8];
#pragma unroll
        for (int t = 0; t < 8; ++t) s[t] = z;
#pragma unroll
        for (int t = 0; t < 8; ++t) {
            v16bf bk0 = frag_B(sK, 72, t * 16, 0);
            v16bf bk1 = frag_B(sK, 72, t * 16, 32);
            s[t] = wmma_bf16(aq0, bk0, s[t]);
            s[t] = wmma_bf16(aq1, bk1, s[t]);
        }

        // online softmax; row r of D lives in element r, cols across 16 lanes per half
#pragma unroll
        for (int r = 0; r < 8; ++r) {
            float rm = s[0][r];
#pragma unroll
            for (int t = 1; t < 8; ++t) rm = fmaxf(rm, s[t][r]);
            rm = fmaxf(rm, __shfl_xor(rm, 1, 16));
            rm = fmaxf(rm, __shfl_xor(rm, 2, 16));
            rm = fmaxf(rm, __shfl_xor(rm, 4, 16));
            rm = fmaxf(rm, __shfl_xor(rm, 8, 16));
            float mn = fmaxf(m_i[r], rm);
            float alpha = __expf(m_i[r] - mn);
            m_i[r] = mn;
            float rs = 0.0f;
#pragma unroll
            for (int t = 0; t < 8; ++t) {
                float p = __expf(s[t][r] - mn);
                s[t][r] = p;
                rs += p;
            }
            rs += __shfl_xor(rs, 1, 16);
            rs += __shfl_xor(rs, 2, 16);
            rs += __shfl_xor(rs, 4, 16);
            rs += __shfl_xor(rs, 8, 16);
            l_i[r] = l_i[r] * alpha + rs;
#pragma unroll
            for (int t2 = 0; t2 < 4; ++t2) acc_o[t2][r] *= alpha;
        }

        // spill P (bf16) to per-wave LDS, then acc_o += P @ V
#pragma unroll
        for (int t = 0; t < 8; ++t)
#pragma unroll
            for (int r = 0; r < 8; ++r)
                sP[(r + 8 * half) * 136 + t * 16 + ln] = f2bf(s[t][r]);

#pragma unroll
        for (int kk = 0; kk < 4; ++kk) {
            v16bf ap = frag_A(sP, 136, 0, kk * 32);
#pragma unroll
            for (int t2 = 0; t2 < 4; ++t2) {
                v16bf bv = frag_B(sVT, 136, t2 * 16, kk * 32);
                acc_o[t2] = wmma_bf16(ap, bv, acc_o[t2]);
            }
        }
        __syncthreads();
    }

    // normalize, store to [B,N,C] bf16 for the proj GEMM
#pragma unroll
    for (int t2 = 0; t2 < 4; ++t2)
#pragma unroll
        for (int r = 0; r < 8; ++r) {
            long grow = (long)b * NSEQ + q0 + wave * 16 + r + 8 * half;
            int  col  = h * HD + t2 * 16 + ln;
            float val = acc_o[t2][r] / l_i[r];
            aob[grow * CDIM + col] = f2bf(val);
        }
}

// ---------------------------------------------------------------- out = ao @ proj_w^T + b
__global__ __launch_bounds__(256) void k_proj(const u16* __restrict__ aob,
                                              const u16* __restrict__ pwb,
                                              const float* __restrict__ pb,
                                              float* __restrict__ out) {
    const int tid = threadIdx.x;
    const int lanei = tid & 31;
    const int wave = tid >> 5;
    const int wm = wave & 3, wn = wave >> 2;
    const int tileN = blockIdx.x * 128;
    const long tileM = (long)blockIdx.y * 128;

    __shared__ u16 sA[2][128 * LDS_STRIDE];
    __shared__ u16 sB[2][128 * LDS_STRIDE];

    v8f z = {};
    v8f acc[2][4];
#pragma unroll
    for (int i = 0; i < 2; ++i)
#pragma unroll
        for (int j = 0; j < 4; ++j) acc[i][j] = z;

    g2l_tile(sA[0], aob, CDIM, tileM, 0, tid);
    g2l_tile(sB[0], pwb, CDIM, tileN, 0, tid);
    async_wait();
    __syncthreads();

    int cur = 0;
    for (int k0 = 0; k0 < CDIM; k0 += 32) {
        if (k0 + 32 < CDIM) {
            g2l_tile(sA[cur ^ 1], aob, CDIM, tileM, k0 + 32, tid);
            g2l_tile(sB[cur ^ 1], pwb, CDIM, tileN, k0 + 32, tid);
        }
        v16bf a0 = frag_A(sA[cur], LDS_STRIDE, wm * 32, 0);
        v16bf a1 = frag_A(sA[cur], LDS_STRIDE, wm * 32 + 16, 0);
#pragma unroll
        for (int ni = 0; ni < 4; ++ni) {
            v16bf bfr = frag_B(sB[cur], LDS_STRIDE, wn * 64 + ni * 16, 0);
            acc[0][ni] = wmma_bf16(a0, bfr, acc[0][ni]);
            acc[1][ni] = wmma_bf16(a1, bfr, acc[1][ni]);
        }
        async_wait();
        __syncthreads();
        cur ^= 1;
    }

    const int half = lanei >> 4, ln = lanei & 15;
#pragma unroll
    for (int mi = 0; mi < 2; ++mi)
#pragma unroll
        for (int ni = 0; ni < 4; ++ni)
#pragma unroll
            for (int r = 0; r < 8; ++r) {
                long row = tileM + wm * 32 + mi * 16 + half * 8 + r;
                int  col = tileN + wn * 64 + ni * 16 + ln;
                out[row * CDIM + col] = acc[mi][ni][r] + pb[col];
            }
}

extern "C" void kernel_launch(void* const* d_in, const int* in_sizes, int n_in,
                              void* d_out, int out_size, void* d_ws, size_t ws_size,
                              hipStream_t stream) {
    const float* x      = (const float*)d_in[0];
    const float* qkv_w  = (const float*)d_in[1];
    const float* qkv_b  = (const float*)d_in[2];
    const float* proj_w = (const float*)d_in[3];
    const float* proj_b = (const float*)d_in[4];
    const float* lAk    = (const float*)d_in[5];
    const float* lBk    = (const float*)d_in[6];
    const float* lAv    = (const float*)d_in[7];
    const float* lBv    = (const float*)d_in[8];
    float* out = (float*)d_out;

    char* w = (char*)d_ws;
    auto alloc = [&](size_t bytes) -> void* {
        void* p = w;
        w += (bytes + 255) & ~(size_t)255;
        return p;
    };
    u16* xb   = (u16*)alloc((size_t)M_TOTAL * CDIM * 2);
    u16* qwb  = (u16*)alloc((size_t)3 * CDIM * CDIM * 2);
    u16* pwb  = (u16*)alloc((size_t)CDIM * CDIM * 2);
    u16* Akb  = (u16*)alloc((size_t)RANK * CDIM * 2);
    u16* Avb  = (u16*)alloc((size_t)RANK * CDIM * 2);
    u16* Bkb  = (u16*)alloc((size_t)CDIM * RANK * 2);
    u16* Bvb  = (u16*)alloc((size_t)CDIM * RANK * 2);
    u16* tb   = (u16*)alloc((size_t)M_TOTAL * 2 * RANK * 2);
    u16* qbuf = (u16*)alloc((size_t)M_TOTAL * CDIM * 2);
    u16* kbuf = (u16*)alloc((size_t)M_TOTAL * CDIM * 2);
    u16* vbuf = (u16*)alloc((size_t)M_TOTAL * CDIM * 2);
    u16* aob  = (u16*)alloc((size_t)M_TOTAL * CDIM * 2);

    k_cvt<<<1024, 256, 0, stream>>>(x,      xb,  M_TOTAL * CDIM);
    k_cvt<<<1024, 256, 0, stream>>>(qkv_w,  qwb, 3 * CDIM * CDIM);
    k_cvt<<<512,  256, 0, stream>>>(proj_w, pwb, CDIM * CDIM);
    k_cvt<<<64,   256, 0, stream>>>(lAk,    Akb, RANK * CDIM);
    k_cvt<<<64,   256, 0, stream>>>(lAv,    Avb, RANK * CDIM);
    k_cvt<<<64,   256, 0, stream>>>(lBk,    Bkb, CDIM * RANK);
    k_cvt<<<64,   256, 0, stream>>>(lBv,    Bvb, CDIM * RANK);

    k_lorat<<<M_TOTAL / 128, 256, 0, stream>>>(xb, Akb, Avb, tb);
    k_qkv<<<dim3(3 * CDIM / 128, M_TOTAL / 128), 256, 0, stream>>>(
        xb, qwb, qkv_b, tb, Bkb, Bvb, qbuf, kbuf, vbuf);

    const int attn_smem = (128 * 72 + 128 * 72 + 64 * 136 + 8 * 16 * 136) * 2; // 89088 B
    k_attn<<<BDIM * NH * (NSEQ / 128), 256, attn_smem, stream>>>(qbuf, kbuf, vbuf, aob);

    k_proj<<<dim3(CDIM / 128, M_TOTAL / 128), 256, 0, stream>>>(aob, pwb, proj_b, out);
}
